// PW_FNet_Block_79894981640571
// MI455X (gfx1250) — matrix operand order repaired and checked by reading, not compiled, not run on hardware
//
#include <hip/hip_runtime.h>
#include <hip/hip_bf16.h>

typedef _Float16 f16;
typedef __attribute__((ext_vector_type(16))) _Float16 v16h;
typedef __attribute__((ext_vector_type(8)))  float    v8f;
typedef __attribute__((ext_vector_type(4)))  float    v4f;
typedef unsigned int u32;
typedef __attribute__((ext_vector_type(4))) u32 v4u;
typedef __attribute__((ext_vector_type(8))) int v8i;
typedef __attribute__((ext_vector_type(4))) int v4i;

#define DIM 64
#define HID 128
#define NPIX (4*256*256)

__device__ __forceinline__ float gelu_f(float v) {
  return 0.5f * v * (1.0f + erff(v * 0.70710678118f));
}
__device__ __forceinline__ v8f wmma32(v16h a, v16h b, v8f c) {
  return __builtin_amdgcn_wmma_f32_16x16x32_f16(false, a, false, b, (short)0, c,
                                                false, false);
}

// Build one A fragment (16x32 f16, ISA 7.12.2 layout) from a row of f32.
// Lane layout: K = half*8 + {0..7} and K = 16 + half*8 + {0..7}, contiguous!
__device__ __forceinline__ v16h loadA(const float* p, int half) {
  const float* q = p + half * 8;
  v4f a0 = *(const v4f*)(q);
  v4f a1 = *(const v4f*)(q + 4);
  v4f a2 = *(const v4f*)(q + 16);
  v4f a3 = *(const v4f*)(q + 20);
  v16h A;
#pragma unroll
  for (int i = 0; i < 4; ++i) {
    A[i]      = (f16)a0[i];
    A[i + 4]  = (f16)a1[i];
    A[i + 8]  = (f16)a2[i];
    A[i + 12] = (f16)a3[i];
  }
  return A;
}

// --- Tensor Data Mover: 2D dword tile, global -> LDS ---------------------
__device__ __forceinline__ void tdm_load_2d(const void* gsrc, u32 lds_byte_off,
                                            u32 dim0_dw, u32 dim1,
                                            u32 stride0_dw) {
#if __has_builtin(__builtin_amdgcn_tensor_load_to_lds)
  unsigned long long ga = (unsigned long long)gsrc;
  v4u g0;
  g0[0] = 1u;                                      // count=1, user mode
  g0[1] = lds_byte_off;                            // lds_addr
  g0[2] = (u32)ga;                                 // global_addr[31:0]
  g0[3] = (u32)((ga >> 32) & 0x01FFFFFFu) | (2u << 30);  // addr[56:32], type=2
  v8i g1;
  g1[0] = (int)(2u << 16);                         // data_size=2 (4B)
  g1[1] = (int)((dim0_dw & 0xFFFFu) << 16);        // tensor_dim0 lo16
  g1[2] = (int)(((dim0_dw >> 16) & 0xFFFFu) | ((dim1 & 0xFFFFu) << 16));
  g1[3] = (int)(((dim1 >> 16) & 0xFFFFu) | ((dim0_dw & 0xFFFFu) << 16)); // tile_dim0
  g1[4] = (int)(dim1 & 0xFFFFu);                   // tile_dim1 (tile_dim2=0)
  g1[5] = (int)stride0_dw;                         // tensor_dim0_stride lo32
  g1[6] = 0;
  g1[7] = 0;
  v4i z4 = {0, 0, 0, 0};
#if defined(__clang_major__) && (__clang_major__ >= 23)
  v8i z8 = {0, 0, 0, 0, 0, 0, 0, 0};
  __builtin_amdgcn_tensor_load_to_lds(g0, g1, z4, z4, z8, 0);
#else
  __builtin_amdgcn_tensor_load_to_lds(g0, g1, z4, z4, 0);
#endif
  __builtin_amdgcn_s_wait_tensorcnt(0);
#else
  (void)gsrc; (void)lds_byte_off; (void)dim0_dw; (void)dim1; (void)stride0_dw;
#endif
}

// --- weight prepack: f32 row-major KxN -> f16 WMMA-B-fragment-major ------
// tile t = nt*(K/32)+kb holds a 32x16 B tile; lane L's 16 halves contiguous:
// dst[t*512 + L*16 + e] = W[(kb*32 + (L>>4)*8 + (e&7) + (e>>3)*16)*N + nt*16 + (L&15)]
__global__ void pack_b(const float* __restrict__ W, f16* __restrict__ dst,
                       int K, int N) {
  int idx = blockIdx.x * 256 + threadIdx.x;
  if (idx >= K * N) return;
  int e    = idx & 15;
  int lane = (idx >> 4) & 31;
  int t    = idx >> 9;
  int nkb  = K >> 5;
  int kb   = t % nkb;
  int nt   = t / nkb;
  int k    = kb * 32 + ((lane >> 4) << 3) + (e & 7) + ((e >> 3) << 4);
  int col  = nt * 16 + (lane & 15);
  dst[idx] = (f16)W[k * N + col];
}

// --- fused LayerNorm(64) + GEMM 64->128 (WMMA) ---------------------------
__global__ void __launch_bounds__(256, 1)
ln_gemm64(const float* __restrict__ in, const float* __restrict__ gam,
          const float* __restrict__ bet, const f16* __restrict__ W,
          const float* __restrict__ bias, float* __restrict__ out) {
  __builtin_prefetch(W, 0, 3);
  int wave = threadIdx.x >> 5, lane = threadIdx.x & 31;
  int row = lane & 15, half = lane >> 4;
  long pixBase = (long)blockIdx.x * 128 + wave * 16;
  const float* px = in + (pixBase + row) * DIM;
  float xr[DIM];
  float mu = 0.f;
#pragma unroll
  for (int c = 0; c < DIM; c += 4) {
    v4f v = *(const v4f*)(px + c);
    xr[c] = v.x; xr[c+1] = v.y; xr[c+2] = v.z; xr[c+3] = v.w;
    mu += v.x + v.y + v.z + v.w;
  }
  mu *= (1.f / DIM);
  float var = 0.f;
#pragma unroll
  for (int c = 0; c < DIM; ++c) { float d = xr[c] - mu; var += d * d; }
  float rs = rsqrtf(var * (1.f / DIM) + 1e-3f);
#pragma unroll
  for (int c = 0; c < DIM; ++c) xr[c] = (xr[c] - mu) * rs * gam[c] + bet[c];
  // A layout: element i of run r: K = r*16 + half*8 + i  (runs at K 0/16/32/48)
  v16h A0, A1;
#pragma unroll
  for (int i = 0; i < 4; ++i) {
    A0[i]      = (f16)xr[half * 8 + i];
    A0[i + 4]  = (f16)xr[half * 8 + 4 + i];
    A0[i + 8]  = (f16)xr[16 + half * 8 + i];
    A0[i + 12] = (f16)xr[16 + half * 8 + 4 + i];
    A1[i]      = (f16)xr[32 + half * 8 + i];
    A1[i + 4]  = (f16)xr[32 + half * 8 + 4 + i];
    A1[i + 8]  = (f16)xr[48 + half * 8 + i];
    A1[i + 12] = (f16)xr[48 + half * 8 + 4 + i];
  }
#pragma unroll 2
  for (int nt = 0; nt < HID / 16; ++nt) {
    int col = nt * 16 + row;
    v16h B0 = *(const v16h*)(W + (nt * 2 + 0) * 512 + lane * 16);
    v16h B1 = *(const v16h*)(W + (nt * 2 + 1) * 512 + lane * 16);
    v8f acc = {};
    acc = wmma32(A0, B0, acc);
    acc = wmma32(A1, B1, acc);
    float bb = bias[col];
#pragma unroll
    for (int v = 0; v < 8; ++v)
      out[(pixBase + v + 8 * half) * HID + col] = acc[v] + bb;
  }
}

// --- GEMM 128->64 + bias + residual (WMMA) -------------------------------
__global__ void __launch_bounds__(256, 1)
res_gemm128(const float* __restrict__ in, const f16* __restrict__ W,
            const float* __restrict__ bias,
            const float* __restrict__ resid, float* __restrict__ out) {
  __builtin_prefetch(W, 0, 3);
  int wave = threadIdx.x >> 5, lane = threadIdx.x & 31;
  int row = lane & 15, half = lane >> 4;
  long pixBase = (long)blockIdx.x * 128 + wave * 16;
  const float* px = in + (pixBase + row) * HID;
  v16h A[4];
#pragma unroll
  for (int kb = 0; kb < 4; ++kb) A[kb] = loadA(px + kb * 32, half);
#pragma unroll 2
  for (int nt = 0; nt < DIM / 16; ++nt) {
    int col = nt * 16 + row;
    v8f acc = {};
#pragma unroll
    for (int kb = 0; kb < 4; ++kb) {
      v16h B = *(const v16h*)(W + (nt * 4 + kb) * 512 + lane * 16);
      acc = wmma32(A[kb], B, acc);
    }
    float bb = bias[col];
#pragma unroll
    for (int v = 0; v < 8; ++v) {
      long o = (pixBase + v + 8 * half) * DIM + col;
      out[o] = acc[v] + bb + resid[o];
    }
  }
}

// --- frequency GEMM 256->256 + GELU; packed w_fc staged in LDS via TDM ---
__global__ void __launch_bounds__(256, 1)
fc_gemm(const float* __restrict__ re, const float* __restrict__ im,
        const f16* __restrict__ Wfc, const float* __restrict__ bias,
        float* __restrict__ outRe, float* __restrict__ outIm) {
  extern __shared__ f16 lds[];  // 256*256 f16 = 128 KB, fragment-major
#if __has_builtin(__builtin_amdgcn_tensor_load_to_lds)
  if ((threadIdx.x >> 5) == 0) {
    tdm_load_2d(Wfc, (u32)(unsigned long long)lds, /*dim0 dwords*/128,
                /*dim1*/256, /*stride dwords*/128);
  }
#else
  for (int i = threadIdx.x; i < 256 * 256; i += 256) lds[i] = Wfc[i];
#endif
  __syncthreads();

  int wave = threadIdx.x >> 5, lane = threadIdx.x & 31;
  int row = lane & 15, half = lane >> 4;
  long pixBase = (long)blockIdx.x * 128 + wave * 16;
  const float* pre = re + (pixBase + row) * HID;
  const float* pim = im + (pixBase + row) * HID;
  v16h A[8];
#pragma unroll
  for (int kb = 0; kb < 4; ++kb) {
    A[kb]     = loadA(pre + kb * 32, half);
    A[kb + 4] = loadA(pim + kb * 32, half);
  }
#pragma unroll 1
  for (int nt = 0; nt < 16; ++nt) {
    int colg = nt * 16 + row;
    v8f acc = {};
#pragma unroll
    for (int kb = 0; kb < 8; ++kb) {
      v16h B = *(const v16h*)(lds + (nt * 8 + kb) * 512 + lane * 16);
      acc = wmma32(A[kb], B, acc);
    }
    float bb = bias[colg];
    float* dst = (colg < HID) ? (outRe + colg) : (outIm + colg - HID);
#pragma unroll
    for (int v = 0; v < 8; ++v) {
      long r = pixBase + v + 8 * half;
      dst[r * HID] = gelu_f(acc[v] + bb);
    }
  }
}

// --- 256-point complex FFT along one axis, four-step (16x16) in LDS ------
// rows: sLo=32768, stride=128 ; cols: sLo=128, stride=32768.
__global__ void fft256(const float* __restrict__ inRe, const float* __restrict__ inIm,
                       float* __restrict__ outRe, float* __restrict__ outIm,
                       int sLo, int stride, float sgn, float scale) {
  __shared__ float r1[4096], i1[4096], r2[4096], i2[4096];
  int blk = blockIdx.x;
  int cg = blk & 7;
  int outer = blk >> 3;
  int b = outer >> 8, lo = outer & 255;
  long base = (long)b * (256L * 256L * 128L) + (long)lo * sLo + cg * 16;

  for (int idx = threadIdx.x; idx < 4096; idx += 256) {
    int line = idx & 15, n = idx >> 4;
    long a = base + line + (long)n * stride;
    r1[line * 256 + n] = inRe[a];
    i1[line * 256 + n] = inIm ? inIm[a] : 0.f;
  }
  __syncthreads();

  int ch = threadIdx.x >> 4, j = threadIdx.x & 15;
  const float w16  = sgn * 0.3926990817f;   // 2*pi/16
  const float w256 = sgn * 0.0245436926f;   // 2*pi/256
  float xr[16], xi[16];
#pragma unroll
  for (int m = 0; m < 16; ++m) {
    xr[m] = r1[ch * 256 + 16 * m + j];
    xi[m] = i1[ch * 256 + 16 * m + j];
  }
#pragma unroll 1
  for (int k1 = 0; k1 < 16; ++k1) {
    float sr = 0.f, si = 0.f;
#pragma unroll
    for (int n1 = 0; n1 < 16; ++n1) {
      float s, c;
      __sincosf(w16 * (float)((n1 * k1) & 15), &s, &c);
      sr += xr[n1] * c - xi[n1] * s;
      si += xi[n1] * c + xr[n1] * s;
    }
    float s, c;
    __sincosf(w256 * (float)(j * k1), &s, &c);
    r2[ch * 256 + j * 16 + k1] = sr * c - si * s;
    i2[ch * 256 + j * 16 + k1] = si * c + sr * s;
  }
  __syncthreads();

  float ar[16], ai[16];
#pragma unroll
  for (int n2 = 0; n2 < 16; ++n2) {
    ar[n2] = r2[ch * 256 + n2 * 16 + j];
    ai[n2] = i2[ch * 256 + n2 * 16 + j];
  }
#pragma unroll 1
  for (int k2 = 0; k2 < 16; ++k2) {
    float sr = 0.f, si = 0.f;
#pragma unroll
    for (int n2 = 0; n2 < 16; ++n2) {
      float s, c;
      __sincosf(w16 * (float)((n2 * k2) & 15), &s, &c);
      sr += ar[n2] * c - ai[n2] * s;
      si += ai[n2] * c + ar[n2] * s;
    }
    r1[ch * 256 + j + 16 * k2] = sr;
    i1[ch * 256 + j + 16 * k2] = si;
  }
  __syncthreads();

  for (int idx = threadIdx.x; idx < 4096; idx += 256) {
    int line = idx & 15, n = idx >> 4;
    long a = base + line + (long)n * stride;
    outRe[a] = r1[line * 256 + n] * scale;
    if (outIm) outIm[a] = i1[line * 256 + n] * scale;
  }
}

// --- depthwise 3x3 SAME conv + bias + GELU, 4 channels / thread ----------
__global__ void dwconv(const float* __restrict__ in, const float* __restrict__ w,
                       const float* __restrict__ bias, float* __restrict__ out) {
  long idx = (long)blockIdx.x * 256 + threadIdx.x;  // 4*256*256*32 threads
  int c4 = (int)(idx & 31);
  int xw = (int)((idx >> 5) & 255);
  int y  = (int)((idx >> 13) & 255);
  int b  = (int)(idx >> 21);
  int c = c4 * 4;
  v4f acc = {0.f, 0.f, 0.f, 0.f};
#pragma unroll
  for (int ky = 0; ky < 3; ++ky) {
    int iy = y + ky - 1;
    if (iy < 0 || iy > 255) continue;
#pragma unroll
    for (int kx = 0; kx < 3; ++kx) {
      int ix = xw + kx - 1;
      if (ix < 0 || ix > 255) continue;
      v4f v  = *(const v4f*)(in + (((long)(b * 256 + iy) * 256 + ix) * HID + c));
      v4f ww = *(const v4f*)(w + ((ky * 3 + kx) * HID + c));
      acc += v * ww;
    }
  }
  acc += *(const v4f*)(bias + c);
  v4f r;
  r.x = gelu_f(acc.x); r.y = gelu_f(acc.y);
  r.z = gelu_f(acc.z); r.w = gelu_f(acc.w);
  *(v4f*)(out + (((long)(b * 256 + y) * 256 + xw) * HID + c)) = r;
}

extern "C" void kernel_launch(void* const* d_in, const int* in_sizes, int n_in,
                              void* d_out, int out_size, void* d_ws, size_t ws_size,
                              hipStream_t stream) {
  (void)in_sizes; (void)n_in; (void)out_size; (void)ws_size;
  const float* x    = (const float*)d_in[0];
  const float* g1v  = (const float*)d_in[1];
  const float* b1v  = (const float*)d_in[2];
  const float* g2v  = (const float*)d_in[3];
  const float* b2v  = (const float*)d_in[4];
  const float* w_te = (const float*)d_in[5];
  const float* b_te = (const float*)d_in[6];
  const float* w_fc = (const float*)d_in[7];
  const float* b_fc = (const float*)d_in[8];
  const float* w_tp = (const float*)d_in[9];
  const float* b_tp = (const float*)d_in[10];
  const float* w_fe = (const float*)d_in[11];
  const float* b_fe = (const float*)d_in[12];
  const float* w_dw = (const float*)d_in[13];
  const float* b_dw = (const float*)d_in[14];
  const float* w_fp = (const float*)d_in[15];
  const float* b_fp = (const float*)d_in[16];

  char* ws = (char*)d_ws;
  f16* wte16 = (f16*)(ws);
  f16* wfc16 = (f16*)(ws + (16 << 10));
  f16* wtp16 = (f16*)(ws + (16 << 10) + (128 << 10));
  f16* wfe16 = (f16*)(ws + (32 << 10) + (128 << 10));
  f16* wfp16 = (f16*)(ws + (48 << 10) + (128 << 10));
  const size_t MB = 1ull << 20;
  float* bufH   = (float*)(ws + 1 * MB);            // 128 MB
  float* bufRe  = (float*)(ws + 1 * MB + 128 * MB); // 128 MB
  float* bufIm  = (float*)(ws + 1 * MB + 256 * MB); // 128 MB
  float* bufRe2 = (float*)(ws + 1 * MB + 384 * MB); // 128 MB
  float* bufIm2 = (float*)(ws + 1 * MB + 512 * MB); // 128 MB
  float* bufX1  = (float*)(ws + 1 * MB + 640 * MB); // 64 MB

  // weight prepack f32 -> f16, WMMA B-fragment-major
  pack_b<<<(DIM * HID + 255) / 256, 256, 0, stream>>>(w_te, wte16, DIM, HID);
  pack_b<<<(256 * 256 + 255) / 256, 256, 0, stream>>>(w_fc, wfc16, 256, 256);
  pack_b<<<(HID * DIM + 255) / 256, 256, 0, stream>>>(w_tp, wtp16, HID, DIM);
  pack_b<<<(DIM * HID + 255) / 256, 256, 0, stream>>>(w_fe, wfe16, DIM, HID);
  pack_b<<<(HID * DIM + 255) / 256, 256, 0, stream>>>(w_fp, wfp16, HID, DIM);

  // h = LN1(x) @ w_te + b_te
  ln_gemm64<<<NPIX / 128, 256, 0, stream>>>(x, g1v, b1v, wte16, b_te, bufH);
  // FFT2 forward: rows then cols
  fft256<<<8192, 256, 0, stream>>>(bufH, nullptr, bufRe, bufIm, 32768, 128, -1.f, 1.f);
  fft256<<<8192, 256, 0, stream>>>(bufRe, bufIm, bufRe2, bufIm2, 128, 32768, -1.f, 1.f);
  // frequency GEMM (256->256) + GELU, packed weights via TDM -> LDS
  fc_gemm<<<NPIX / 128, 256, 256 * 256 * sizeof(f16), stream>>>(
      bufRe2, bufIm2, wfc16, b_fc, bufRe, bufIm);
  // inverse FFT2: cols then rows (1/256 per stage), real part -> sp
  fft256<<<8192, 256, 0, stream>>>(bufRe, bufIm, bufRe2, bufIm2, 128, 32768, 1.f, 1.f / 256.f);
  fft256<<<8192, 256, 0, stream>>>(bufRe2, bufIm2, bufH, nullptr, 32768, 128, 1.f, 1.f / 256.f);
  // x1 = x + sp @ w_tp + b_tp
  res_gemm128<<<NPIX / 128, 256, 0, stream>>>(bufH, wtp16, b_tp, x, bufX1);
  // h2 = LN2(x1) @ w_fe + b_fe
  ln_gemm64<<<NPIX / 128, 256, 0, stream>>>(bufX1, g2v, b2v, wfe16, b_fe, bufRe);
  // d = gelu(dwconv3x3(h2) + b_dw)
  dwconv<<<(NPIX * HID / 4) / 256, 256, 0, stream>>>(bufRe, w_dw, b_dw, bufIm);
  // out = x1 + d @ w_fp + b_fp
  res_gemm128<<<NPIX / 128, 256, 0, stream>>>(bufIm, wfp16, b_fp, bufX1, (float*)d_out);
}
